// Attention_51831665328669
// MI455X (gfx1250) — compile-verified, matrix operand order checked by reading, not run on hardware
//
#include <hip/hip_runtime.h>

typedef __attribute__((ext_vector_type(16))) __bf16 v16bf;
typedef __attribute__((ext_vector_type(8)))  __bf16 v8bf;
typedef __attribute__((ext_vector_type(4)))  __bf16 v4bf;
typedef __attribute__((ext_vector_type(8)))  float  v8f;

namespace {
constexpr int Sseq = 2048;
constexpr int Dh   = 64;
constexpr int MT   = 128;  // q rows per block
constexpr int KC   = 32;   // keys per chunk
constexpr int NW   = 8;    // waves per block (256 threads, wave32)
constexpr int NT   = NW * 32;
// 1/sqrt(64) * log2(e): softmax in base-2 so exp is a single v_exp_f32
__device__ constexpr float QSCALE = 0.125f * 1.4426950408889634f;
}

union ABf { v16bf v; v8bf h[2]; };

// 16-lane butterfly reductions done purely in VALU via DPP16 permutations
// (0xB1 quad_perm[1,0,3,2], 0x4E quad_perm[2,3,0,1], 0x141 row_half_mirror,
//  0x140 row_mirror) — each involution combines disjoint halves of the row.
__device__ __forceinline__ float row16_max(float x) {
  union { float f; int i; } a, b;
  a.f = x;
  b.i = __builtin_amdgcn_update_dpp(0, a.i, 0xB1,  0xF, 0xF, true);
  a.f = fmaxf(a.f, b.f);
  b.i = __builtin_amdgcn_update_dpp(0, a.i, 0x4E,  0xF, 0xF, true);
  a.f = fmaxf(a.f, b.f);
  b.i = __builtin_amdgcn_update_dpp(0, a.i, 0x141, 0xF, 0xF, true);
  a.f = fmaxf(a.f, b.f);
  b.i = __builtin_amdgcn_update_dpp(0, a.i, 0x140, 0xF, 0xF, true);
  a.f = fmaxf(a.f, b.f);
  return a.f;
}
__device__ __forceinline__ float row16_sum(float x) {
  union { float f; int i; } a, b;
  a.f = x;
  b.i = __builtin_amdgcn_update_dpp(0, a.i, 0xB1,  0xF, 0xF, true);
  a.f += b.f;
  b.i = __builtin_amdgcn_update_dpp(0, a.i, 0x4E,  0xF, 0xF, true);
  a.f += b.f;
  b.i = __builtin_amdgcn_update_dpp(0, a.i, 0x141, 0xF, 0xF, true);
  a.f += b.f;
  b.i = __builtin_amdgcn_update_dpp(0, a.i, 0x140, 0xF, 0xF, true);
  a.f += b.f;
  return a.f;
}

__global__ __launch_bounds__(NT, 1) void fa_fwd_kernel(
    const float* __restrict__ Q, const float* __restrict__ K,
    const float* __restrict__ V, float* __restrict__ O)
{
  // double-buffered fp32 staging filled by async DMA, plus bf16 tile buffers
  __shared__ alignas(16) float  stgK[2][KC][Dh];   // 2 x 8KB
  __shared__ alignas(16) float  stgV[2][KC][Dh];   // 2 x 8KB
  __shared__ alignas(16) __bf16 Ks[KC][Dh];        // K chunk, [key][d]
  __shared__ alignas(16) __bf16 Vt[Dh][KC];        // V chunk transposed, [d][key]
  __shared__ alignas(16) __bf16 Ps[NW][16][KC];    // per-wave P staging (C->A)

  const int tid  = threadIdx.x;
  const int wav  = tid >> 5;
  const int lane = tid & 31;
  const int hl   = lane & 15;
  const int hi   = lane >> 4;

  const int q0 = blockIdx.x * MT;
  const int bh = blockIdx.y;                 // b*H + h
  const int qw = q0 + wav * 16;              // this wave's first q row
  const int qw_u = __builtin_amdgcn_readfirstlane(qw);  // scalar causal bound

  const float* Qb = Q + (size_t)bh * Sseq * Dh;
  const float* Kb = K + (size_t)bh * Sseq * Dh;
  const float* Vb = V + (size_t)bh * Sseq * Dh;
  float*       Ob = O + (size_t)bh * Sseq * Dh;

  // ---- Q rows into WMMA A-layout registers (scaled) ----
  // A 16x32 bf16: lane holds row M=hl; lanes<16 K={0..7,16..23}, lanes>=16 K={8..15,24..31}
  ABf Qa[2];
  {
    const float* qp = Qb + (size_t)(qw + hl) * Dh;
    #pragma unroll
    for (int t = 0; t < 2; ++t)
      #pragma unroll
      for (int c = 0; c < 2; ++c) {
        const int d0 = 32*t + 16*c + 8*hi;
        #pragma unroll
        for (int e = 0; e < 8; ++e)
          Qa[t].h[c][e] = (__bf16)(qp[d0 + e] * QSCALE);
      }
  }

  v8f Oacc[4];
  #pragma unroll
  for (int t = 0; t < 4; ++t)
    #pragma unroll
    for (int e = 0; e < 8; ++e) Oacc[t][e] = 0.0f;

  float mrow[8], lrow[8];
  #pragma unroll
  for (int r = 0; r < 8; ++r) { mrow[r] = -1e30f; lrow[r] = 0.0f; }

  // issue one K/V chunk into staging buffer `buf` via async DMA to LDS
  auto issue_async = [&](int kbase, int buf) {
    #pragma unroll
    for (int i = 0; i < 2; ++i) {
      const int f = tid + NT * i;            // float4 index, 0..511
      const float* gk = Kb + (size_t)kbase * Dh + f * 4;
      const float* gv = Vb + (size_t)kbase * Dh + f * 4;
      const unsigned lk = (unsigned)(uintptr_t)&stgK[buf][0][0] + f * 16;
      const unsigned lv = (unsigned)(uintptr_t)&stgV[buf][0][0] + f * 16;
      asm volatile("global_load_async_to_lds_b128 %0, %1, off"
                   :: "v"(lk), "v"((unsigned long long)(uintptr_t)gk) : "memory");
      asm volatile("global_load_async_to_lds_b128 %0, %1, off"
                   :: "v"(lv), "v"((unsigned long long)(uintptr_t)gv) : "memory");
    }
  };

  const int nchunks = (q0 + MT) / KC;        // causal bound for this block
  int cur = 0;
  issue_async(0, 0);

  for (int kc = 0; kc < nchunks; ++kc) {
    const int kbase = kc * KC;
    // issue next chunk first, then wait only for the oldest 4 async ops
    // (async loads complete in order; double-buffer reuse is fenced by the
    //  previous iteration's post-convert barrier)
    if (kc + 1 < nchunks) {
      issue_async(kbase + KC, cur ^ 1);
      asm volatile("s_wait_asynccnt 0x4" ::: "memory");
    } else {
      asm volatile("s_wait_asynccnt 0x0" ::: "memory");
    }
    __syncthreads();                          // staging[cur] visible to all waves

    // ---- cooperative fp32->bf16 tile build from staging (V transposed) ----
    #pragma unroll
    for (int i = 0; i < 2; ++i) {
      const int f   = tid + NT * i;
      const int key = f >> 4;
      const int dp  = (f & 15) << 2;
      const float4 kq = *(const float4*)&stgK[cur][key][dp];
      v4bf kb;
      kb[0] = (__bf16)kq.x; kb[1] = (__bf16)kq.y;
      kb[2] = (__bf16)kq.z; kb[3] = (__bf16)kq.w;
      *(v4bf*)&Ks[key][dp] = kb;
      const float4 vv = *(const float4*)&stgV[cur][key][dp];
      Vt[dp+0][key] = (__bf16)vv.x; Vt[dp+1][key] = (__bf16)vv.y;
      Vt[dp+2][key] = (__bf16)vv.z; Vt[dp+3][key] = (__bf16)vv.w;
    }
    __syncthreads();

    if (kbase <= qw_u + 15) {      // scalar causal skip -> EXEC stays all-1s
      // ---- scores: S(16x32) = Q(16x64) x K^T, 4 WMMAs ----
      v8f s[2];
      #pragma unroll
      for (int sg = 0; sg < 2; ++sg) {
        #pragma unroll
        for (int e = 0; e < 8; ++e) s[sg][e] = 0.0f;
        #pragma unroll
        for (int t = 0; t < 2; ++t) {
          ABf Bk;   // B 32x16: lane = key column, K(=d) split like A layout
          Bk.h[0] = *(const v8bf*)&Ks[16*sg + hl][32*t + 8*hi];
          Bk.h[1] = *(const v8bf*)&Ks[16*sg + hl][32*t + 16 + 8*hi];
          s[sg] = __builtin_amdgcn_wmma_f32_16x16x32_bf16(
              false, Qa[t].v, false, Bk.v, (short)0, s[sg], false, false);
        }
      }
      // ---- causal mask only on diagonal chunks (uniform test) ----
      if (kbase + KC - 1 > qw_u) {
        const int qr0 = qw + 8*hi;
        #pragma unroll
        for (int sg = 0; sg < 2; ++sg) {
          const int key = kbase + 16*sg + hl;
          #pragma unroll
          for (int r = 0; r < 8; ++r)
            if (key > qr0 + r) s[sg][r] = -1e30f;
        }
      }
      // ---- online softmax (base-2), reductions in VALU via DPP16 ----
      float al[8];
      #pragma unroll
      for (int r = 0; r < 8; ++r) {
        const float mc = row16_max(fmaxf(s[0][r], s[1][r]));
        const float mn = fmaxf(mrow[r], mc);
        al[r] = __builtin_amdgcn_exp2f(mrow[r] - mn);
        mrow[r] = mn;
      }
      #pragma unroll
      for (int r = 0; r < 8; ++r) {
        const float p0 = __builtin_amdgcn_exp2f(s[0][r] - mrow[r]);
        const float p1 = __builtin_amdgcn_exp2f(s[1][r] - mrow[r]);
        s[0][r] = p0; s[1][r] = p1;
        lrow[r] = lrow[r] * al[r] + row16_sum(p0 + p1);
      }
      #pragma unroll
      for (int t = 0; t < 4; ++t)
        #pragma unroll
        for (int r = 0; r < 8; ++r) Oacc[t][r] *= al[r];

      // ---- stage P through LDS: C-layout -> A-layout ----
      #pragma unroll
      for (int sg = 0; sg < 2; ++sg)
        #pragma unroll
        for (int r = 0; r < 8; ++r)
          Ps[wav][r + 8*hi][16*sg + hl] = (__bf16)s[sg][r];
      __asm__ volatile("" ::: "memory");   // same-wave DS RAW: keep order

      ABf Pa;
      Pa.h[0] = *(const v8bf*)&Ps[wav][hl][8*hi];
      Pa.h[1] = *(const v8bf*)&Ps[wav][hl][16 + 8*hi];

      // ---- O(16x64) += P(16x32) x V(32x64): 4 WMMAs ----
      #pragma unroll
      for (int t = 0; t < 4; ++t) {
        ABf Vw;   // B 32x16: lane = d column (transposed V), K = key
        Vw.h[0] = *(const v8bf*)&Vt[16*t + hl][8*hi];
        Vw.h[1] = *(const v8bf*)&Vt[16*t + hl][16 + 8*hi];
        Oacc[t] = __builtin_amdgcn_wmma_f32_16x16x32_bf16(
            false, Pa.v, false, Vw.v, (short)0, Oacc[t], false, false);
      }
    }
    cur ^= 1;
  }

  // ---- normalize and store (C layout: row = qw + 8*hi + r, col = 16*t + hl) ----
  const int orow0 = qw + 8*hi;
  #pragma unroll
  for (int r = 0; r < 8; ++r) {
    const float inv = 1.0f / lrow[r];
    #pragma unroll
    for (int t = 0; t < 4; ++t)
      Ob[(size_t)(orow0 + r) * Dh + 16*t + hl] = Oacc[t][r] * inv;
  }
}

extern "C" void kernel_launch(void* const* d_in, const int* in_sizes, int n_in,
                              void* d_out, int out_size, void* d_ws, size_t ws_size,
                              hipStream_t stream) {
  (void)in_sizes; (void)n_in; (void)out_size; (void)d_ws; (void)ws_size;
  const float* q = (const float*)d_in[0];
  const float* k = (const float*)d_in[1];
  const float* v = (const float*)d_in[2];
  // d_in[3] is the triu mask; causality is computed analytically in-kernel.
  dim3 grid(Sseq / MT, 4 * 16);   // 16 q-tiles x (B*H)
  fa_fwd_kernel<<<grid, dim3(NT), 0, stream>>>(q, k, v, (float*)d_out);
}
